// CompressKV_88235808129249
// MI455X (gfx1250) — matrix operand order repaired and checked by reading, not compiled, not run on hardware
//
#include <hip/hip_runtime.h>
#include <math.h>

typedef __attribute__((ext_vector_type(16))) _Float16 v16h;
typedef __attribute__((ext_vector_type(8)))  _Float16 v8h;
typedef __attribute__((ext_vector_type(8)))  float    v8f;

#define KSZ   32
#define STR   16
#define B_    2
#define N_    4096
#define H_    16
#define D_    128
#define NBLK  ((N_ - KSZ) / STR + 1)   // 255
#define KD    (KSZ * D_)               // 4096 flattened reduction

#define CVT4(v, base, q)                       \
  v[(base)+0] = (_Float16)(q).x;               \
  v[(base)+1] = (_Float16)(q).y;               \
  v[(base)+2] = (_Float16)(q).z;               \
  v[(base)+3] = (_Float16)(q).w;

__global__ __launch_bounds__(256)
void compress_gate_kernel(const float* __restrict__ x,
                          const float* __restrict__ Wg,
                          float* __restrict__ out)
{
    // 128 KB f16 window tile, 16 KB wave-partial logits, 2 KB logits/gates
    __shared__ __align__(32) _Float16 tile[KSZ][H_][D_];
    __shared__ __align__(32) float    part[8][H_][KSZ];   // [wave][h][g]
    __shared__ float gl[KSZ][H_];                         // logits -> gates

    const int blk  = blockIdx.x;          // 0 .. B_*NBLK-1
    const int b    = blk / NBLK;
    const int nblk = blk - b * NBLK;
    const int n0   = nblk * STR;

    const int tid  = threadIdx.x;
    const int wave = tid >> 5;            // 8 waves (wave32)
    const int lane = tid & 31;
    const int hi   = lane >> 4;           // 0: lanes 0-15, 1: lanes 16-31
    const int l15  = lane & 15;

    // ---------------- Phase 1: gate GEMM via WMMA ----------------
    // logits(32g x 16h) = W(32g x 4096kd) * X(4096kd x 16h)
    // wave w owns K range [w*512, w*512+512): 16 k-steps of K=32.
    v8f c0 = {};  // g = 0..15 tile
    v8f c1 = {};  // g = 16..31 tile

    const float* Wrow0 = Wg + (size_t)l15 * KD;          // A row for tile 0
    const float* Wrow1 = Wrow0 + (size_t)16 * KD;        // A row for tile 1
    const int koA = hi ? 16 : 0;  // A-frag relative K: {0..15} or {16..31}

#pragma unroll 2
    for (int t = 0; t < 16; ++t) {
        const int kd    = (wave * 16 + t) * 32;  // flattened K base = k*128 + d
        const int krow  = kd >> 7;               // window row 0..31
        const int dbase = kd & 127;              // 0,32,64,96
        const int dofs  = dbase + (hi << 4);     // this half-wave's 16 d's

        // ----- B fragment: X[kd+0..31, head=l15]; lanes hold 16 contiguous d -----
        const float4* xp = reinterpret_cast<const float4*>(
            x + ((((size_t)b * N_ + (n0 + krow)) * H_ + l15) * D_ + dofs));
        float4 q0 = xp[0], q1 = xp[1], q2 = xp[2], q3 = xp[3];
        v16h bf;
        CVT4(bf, 0,  q0); CVT4(bf, 4,  q1); CVT4(bf, 8,  q2); CVT4(bf, 12, q3);

        // stage into LDS tile (collectively the 8 waves cover the whole window)
        *reinterpret_cast<v16h*>(&tile[krow][l15][dofs]) = bf;

        // ----- A fragments: W rows, two contiguous 8-float chunks -----
        const float4* w0a = reinterpret_cast<const float4*>(Wrow0 + kd + koA);
        const float4* w0b = reinterpret_cast<const float4*>(Wrow0 + kd + 8 + koA);
        float4 p0 = w0a[0], p1 = w0a[1], p2 = w0b[0], p3 = w0b[1];
        v16h a0;
        CVT4(a0, 0,  p0); CVT4(a0, 4,  p1); CVT4(a0, 8,  p2); CVT4(a0, 12, p3);

        const float4* w1a = reinterpret_cast<const float4*>(Wrow1 + kd + koA);
        const float4* w1b = reinterpret_cast<const float4*>(Wrow1 + kd + 8 + koA);
        float4 r0 = w1a[0], r1 = w1a[1], r2 = w1b[0], r3 = w1b[1];
        v16h a1;
        CVT4(a1, 0,  r0); CVT4(a1, 4,  r1); CVT4(a1, 8,  r2); CVT4(a1, 12, r3);

        c0 = __builtin_amdgcn_wmma_f32_16x16x32_f16(false, a0, false, bf,
                                                    (short)0, c0, false, false);
        c1 = __builtin_amdgcn_wmma_f32_16x16x32_f16(false, a1, false, bf,
                                                    (short)0, c1, false, false);
    }

    // spill per-wave partial logits, vectorized:
    // C/D layout -> c0[r] is (g = hi*8 + r, h = l15); 8 contiguous g per store.
    {
        const int gb = hi * 8;
        *reinterpret_cast<v8f*>(&part[wave][l15][gb])      = c0;  // g in [gb, gb+8)
        *reinterpret_cast<v8f*>(&part[wave][l15][16 + gb]) = c1;  // g in [16+gb, ..)
    }
    __syncthreads();

    // ---------------- Phase 2: reduce partials + softmax over g ----------------
#pragma unroll
    for (int j = 0; j < 2; ++j) {
        const int e = tid * 2 + j;        // 512 (g,h) entries
        const int g = e & 31, h = e >> 5;
        float s = 0.f;
#pragma unroll
        for (int w = 0; w < 8; ++w) s += part[w][h][g];
        gl[g][h] = s;
    }
    __syncthreads();

    if (tid < H_) {
        const int h = tid;
        float m = -INFINITY;
        for (int g = 0; g < KSZ; ++g) m = fmaxf(m, gl[g][h]);
        float s = 0.f;
        for (int g = 0; g < KSZ; ++g) {
            float e = __expf(gl[g][h] - m);
            gl[g][h] = e;
            s += e;
        }
        const float inv = 1.f / s;
        for (int g = 0; g < KSZ; ++g) gl[g][h] *= inv;
    }
    __syncthreads();

    // ------- Phase 3: out[h,d] = sum_k gate[k,h] * tile[k,h,d] -------
    // Thread owns head h3 and 8 contiguous d: one ds_load_b128 per k.
    {
        const int h3 = tid >> 4;           // 0..15
        const int d0 = (tid & 15) * 8;     // 0,8,...,120
        float acc[8] = {0.f, 0.f, 0.f, 0.f, 0.f, 0.f, 0.f, 0.f};
#pragma unroll
        for (int k = 0; k < KSZ; ++k) {
            const float gk = gl[k][h3];
            const v8h v = *reinterpret_cast<const v8h*>(&tile[k][h3][d0]);
#pragma unroll
            for (int j = 0; j < 8; ++j) acc[j] += gk * (float)v[j];
        }
        float* op = out + (((size_t)b * NBLK + nblk) * H_ + h3) * D_ + d0;
#pragma unroll
        for (int j = 0; j < 8; ++j) op[j] = acc[j];
    }
}

extern "C" void kernel_launch(void* const* d_in, const int* in_sizes, int n_in,
                              void* d_out, int out_size, void* d_ws, size_t ws_size,
                              hipStream_t stream) {
    const float* x  = (const float*)d_in[0];   // (2, 4096, 16, 128) f32
    const float* Wg = (const float*)d_in[1];   // (32, 4096) f32
    float* out = (float*)d_out;                // (2, 255, 16, 128) f32

    (void)in_sizes; (void)n_in; (void)out_size; (void)d_ws; (void)ws_size;

    dim3 grid(B_ * NBLK);   // 510 window tiles
    dim3 block(256);        // 8 wave32 waves
    compress_gate_kernel<<<grid, block, 0, stream>>>(x, Wg, out);
}
